// Phi3Attention_32899449487722
// MI455X (gfx1250) — compile-verified, matrix operand order checked by reading, not run on hardware
//
#include <hip/hip_runtime.h>

// ---------------------------------------------------------------------------
// Types for gfx1250 WMMA (wave32): A/B = 16 bf16 per lane, C/D = 8 f32 per lane
// ---------------------------------------------------------------------------
typedef __attribute__((ext_vector_type(16))) __bf16 bf16x16;
typedef __attribute__((ext_vector_type(8)))  float  f32x8;
typedef __attribute__((ext_vector_type(4)))  int    v4i;

struct FragBits { uint4 lo, hi; };   // 32 bytes == bf16x16

__device__ __forceinline__ unsigned short f2bf_u(float f) {
    unsigned u = __builtin_bit_cast(unsigned, f);
    u += 0x7FFFu + ((u >> 16) & 1u);          // round-to-nearest-even
    return (unsigned short)(u >> 16);
}
__device__ __forceinline__ __bf16 f2bf16(float f) {
    unsigned short s = f2bf_u(f);
    return __builtin_bit_cast(__bf16, s);
}
__device__ __forceinline__ unsigned pk2(float a, float b) {
    return (unsigned)f2bf_u(a) | ((unsigned)f2bf_u(b) << 16);
}

// Load one K=32 chunk of an A/B fragment (ISA 7.12.2 16-bit A 16x32 layout).
__device__ __forceinline__ bf16x16 ld_chunk(const __bf16* rowp, int c, int hf) {
    FragBits fb;
    fb.lo = *(const uint4*)(rowp + c * 32 + hf * 8);
    fb.hi = *(const uint4*)(rowp + c * 32 + 16 + hf * 8);
    return __builtin_bit_cast(bf16x16, fb);
}

// ---------------------------------------------------------------------------
// gfx1250 async global->LDS copies (ASYNCcnt-tracked), guarded so compile
// cannot regress if the builtin is absent on this toolchain.
// Signature (from hipcc diagnostic): (v4i AS1*, v4i AS3*, imm offset, imm cpol)
// ---------------------------------------------------------------------------
#if defined(__HIP_DEVICE_COMPILE__) && \
    __has_builtin(__builtin_amdgcn_global_load_async_to_lds_b128)
#define USE_ASYNC_LDS 1
#else
#define USE_ASYNC_LDS 0
#endif

#if USE_ASYNC_LDS
typedef __attribute__((address_space(1))) v4i gv4i;
typedef __attribute__((address_space(3))) v4i lv4i;
__device__ __forceinline__ void async_cp_b128(const void* g, void* l) {
    __builtin_amdgcn_global_load_async_to_lds_b128((gv4i*)g, (lv4i*)l, 0, 0);
}
__device__ __forceinline__ void wait_async0() {
#if __has_builtin(__builtin_amdgcn_s_wait_asynccnt)
    __builtin_amdgcn_s_wait_asynccnt(0);
#else
    asm volatile("s_wait_asynccnt 0" ::: "memory");
#endif
}
#endif

// ---------------------------------------------------------------------------
// Tiled GEMM: C[MrxN] = A[MrxK] * B[KxN], fp32 accumulate via bf16 WMMA.
// Double-buffered LDS software pipeline: next tile's global loads are issued
// before the WMMA block of the current tile. A is fp32 (converted on stage-in)
// or already bf16 (then staged with async global->LDS DMA when available).
// Block 256 thr = 8 waves; BM=128,BN=128,BK=32; wave computes 64x32.
// ---------------------------------------------------------------------------
template <bool ABF16>
__global__ void __launch_bounds__(256)
wmma_gemm(const void* __restrict__ Ap, const float* __restrict__ B,
          float* __restrict__ C, int Mr, int N, int K) {
    constexpr int BM = 128, BN = 128, BK = 32, LDA = 40;  // 80B rows, 16B aligned
    __shared__ __align__(16) __bf16 As[2][BM][LDA];
    __shared__ __align__(16) __bf16 Bs[2][BN][LDA];       // Bs[n][k] = B^T tile

    const int tid = threadIdx.x;
    const int lane = tid & 31, wid = tid >> 5;
    const int hf = lane >> 4, l16 = lane & 15;
    const int m0 = blockIdx.y * BM, n0 = blockIdx.x * BN;
    const int wm = (wid & 1) * 64;      // wave row block
    const int wn = (wid >> 1) * 32;     // wave col block

    const int am = tid >> 1, ak = (tid & 1) * 16;   // A staging: 16 halfs / thread
    const int bk = tid >> 3, bn = (tid & 7) * 16;   // B staging: 16 cols / thread

    const float*  Af = (const float*)Ap;
    const __bf16* Ab = (const __bf16*)Ap;

    f32x8 acc[4][2] = {};
    const int niter = K / BK;

    // ---- prologue: stage tile 0 into buffer 0
    if (ABF16) {
#if USE_ASYNC_LDS
        const __bf16* src = Ab + (size_t)(m0 + am) * K + ak;
        async_cp_b128(src,     &As[0][am][ak]);
        async_cp_b128(src + 8, &As[0][am][ak + 8]);
#else
        const uint4* src = (const uint4*)(Ab + (size_t)(m0 + am) * K + ak);
        *(uint4*)&As[0][am][ak]     = src[0];
        *(uint4*)&As[0][am][ak + 8] = src[1];
#endif
    } else {
        const float4* src = (const float4*)(Af + (size_t)(m0 + am) * K + ak);
        float4 f0 = src[0], f1 = src[1], f2 = src[2], f3 = src[3];
        uint4 u0 = { pk2(f0.x, f0.y), pk2(f0.z, f0.w), pk2(f1.x, f1.y), pk2(f1.z, f1.w) };
        uint4 u1 = { pk2(f2.x, f2.y), pk2(f2.z, f2.w), pk2(f3.x, f3.y), pk2(f3.z, f3.w) };
        *(uint4*)&As[0][am][ak]     = u0;
        *(uint4*)&As[0][am][ak + 8] = u1;
    }
    {
        const float4* src = (const float4*)(B + (size_t)bk * N + n0 + bn);
        float4 g0 = src[0], g1 = src[1], g2 = src[2], g3 = src[3];
        float v[16] = { g0.x, g0.y, g0.z, g0.w, g1.x, g1.y, g1.z, g1.w,
                        g2.x, g2.y, g2.z, g2.w, g3.x, g3.y, g3.z, g3.w };
#pragma unroll
        for (int j = 0; j < 16; ++j) Bs[0][bn + j][bk] = f2bf16(v[j]);
    }
#if USE_ASYNC_LDS
    if (ABF16) wait_async0();
#endif
    __syncthreads();

    for (int it = 0; it < niter; ++it) {
        const int cur = it & 1, nxt = cur ^ 1;
        const bool hn = (it + 1) < niter;
        const int kb2 = (it + 1) * BK;

        // ---- issue next-tile global traffic before the WMMA block
        float4 fa0, fa1, fa2, fa3, gb0, gb1, gb2, gb3;
        uint4 ua0, ua1;
        if (hn) {
            if (ABF16) {
#if USE_ASYNC_LDS
                const __bf16* src = Ab + (size_t)(m0 + am) * K + kb2 + ak;
                async_cp_b128(src,     &As[nxt][am][ak]);
                async_cp_b128(src + 8, &As[nxt][am][ak + 8]);
#else
                const uint4* src = (const uint4*)(Ab + (size_t)(m0 + am) * K + kb2 + ak);
                ua0 = src[0]; ua1 = src[1];
#endif
            } else {
                const float4* src = (const float4*)(Af + (size_t)(m0 + am) * K + kb2 + ak);
                fa0 = src[0]; fa1 = src[1]; fa2 = src[2]; fa3 = src[3];
            }
            const float4* sb = (const float4*)(B + (size_t)(kb2 + bk) * N + n0 + bn);
            gb0 = sb[0]; gb1 = sb[1]; gb2 = sb[2]; gb3 = sb[3];
        }
        if (it + 2 < niter) {   // L2 prefetch two tiles ahead (global_prefetch_b8)
            __builtin_prefetch(B + (size_t)((it + 2) * BK + bk) * N + n0 + bn, 0, 1);
            if (!ABF16)
                __builtin_prefetch(Af + (size_t)(m0 + am) * K + (it + 2) * BK + ak, 0, 1);
        }

        // ---- compute current tile
        bf16x16 afr[4], bfr[2];
#pragma unroll
        for (int mt = 0; mt < 4; ++mt) afr[mt] = ld_chunk(&As[cur][wm + mt * 16 + l16][0], 0, hf);
#pragma unroll
        for (int nt = 0; nt < 2; ++nt) bfr[nt] = ld_chunk(&Bs[cur][wn + nt * 16 + l16][0], 0, hf);
#pragma unroll
        for (int mt = 0; mt < 4; ++mt)
#pragma unroll
            for (int nt = 0; nt < 2; ++nt)
                acc[mt][nt] = __builtin_amdgcn_wmma_f32_16x16x32_bf16(
                    false, afr[mt], false, bfr[nt], (short)0, acc[mt][nt], false, false);

        // ---- convert/store next tile into the other buffer
        if (hn) {
            if (ABF16) {
#if !USE_ASYNC_LDS
                *(uint4*)&As[nxt][am][ak]     = ua0;
                *(uint4*)&As[nxt][am][ak + 8] = ua1;
#endif
            } else {
                uint4 u0 = { pk2(fa0.x, fa0.y), pk2(fa0.z, fa0.w), pk2(fa1.x, fa1.y), pk2(fa1.z, fa1.w) };
                uint4 u1 = { pk2(fa2.x, fa2.y), pk2(fa2.z, fa2.w), pk2(fa3.x, fa3.y), pk2(fa3.z, fa3.w) };
                *(uint4*)&As[nxt][am][ak]     = u0;
                *(uint4*)&As[nxt][am][ak + 8] = u1;
            }
            float v[16] = { gb0.x, gb0.y, gb0.z, gb0.w, gb1.x, gb1.y, gb1.z, gb1.w,
                            gb2.x, gb2.y, gb2.z, gb2.w, gb3.x, gb3.y, gb3.z, gb3.w };
#pragma unroll
            for (int j = 0; j < 16; ++j) Bs[nxt][bn + j][bk] = f2bf16(v[j]);
#if USE_ASYNC_LDS
            if (ABF16) wait_async0();
#endif
        }
        __syncthreads();
    }

    // ---- epilogue: C/D layout => VGPR r, lanes 0-15: M=r, lanes 16-31: M=r+8
#pragma unroll
    for (int mt = 0; mt < 4; ++mt)
#pragma unroll
        for (int nt = 0; nt < 2; ++nt)
#pragma unroll
            for (int r = 0; r < 8; ++r) {
                int gm = m0 + wm + mt * 16 + hf * 8 + r;
                int gn = n0 + wn + nt * 16 + l16;
                C[(size_t)gm * N + gn] = acc[mt][nt][r];
            }
    (void)Mr;
}

// ---------------------------------------------------------------------------
// RoPE + fp32->bf16 + transpose to head-major [head][S][96]
// ---------------------------------------------------------------------------
__global__ void rope_to_bf16(const float* __restrict__ X, __bf16* __restrict__ Y,
                             int heads, int S) {
    int idx = blockIdx.x * blockDim.x + threadIdx.x;
    int total = S * heads * 48;
    if (idx >= total) return;
    int h = idx % 48;
    int u = (idx / 48) % heads;
    int s = idx / (48 * heads);
    const float LN1E4_48 = 0.19188207734926543f;  // ln(10000)/48
    float ang = (float)s * __expf(-(float)h * LN1E4_48);
    float c = __cosf(ang), sn = __sinf(ang);
    size_t in = (size_t)s * heads * 96 + (size_t)u * 96 + h;
    float x1 = X[in], x2 = X[in + 48];
    size_t o = ((size_t)u * S + s) * 96 + h;
    Y[o]      = f2bf16(x1 * c - x2 * sn);
    Y[o + 48] = f2bf16(x2 * c + x1 * sn);
}

__global__ void conv_to_bf16_hm(const float* __restrict__ X, __bf16* __restrict__ Y,
                                int heads, int S) {
    int idx = blockIdx.x * blockDim.x + threadIdx.x;
    int total = S * heads * 96;
    if (idx >= total) return;
    int h = idx % 96;
    int u = (idx / 96) % heads;
    int s = idx / (96 * heads);
    Y[((size_t)u * S + s) * 96 + h] = f2bf16(X[(size_t)s * heads * 96 + u * 96 + h]);
}

// ---------------------------------------------------------------------------
// Flash attention: block = (q-tile of 128 rows, head u); kv head v = u/4.
// 8 waves x 16 q-rows. 64-key tiles; online softmax; WMMA for QK^T and PV.
// K tile staged via async global->LDS DMA when available.
// Output bf16 at [s][u*96+h] (row stride 3072) to feed the Wo GEMM.
// ---------------------------------------------------------------------------
__global__ void __launch_bounds__(256)
flash_attn(const __bf16* __restrict__ Q, const __bf16* __restrict__ Kh,
           const __bf16* __restrict__ Vh, __bf16* __restrict__ O, int S) {
    constexpr int BR = 128, KT = 64, H = 96;
    __shared__ __align__(16) __bf16 Ks[KT][H];        // key tile, row-major
    __shared__ __align__(16) __bf16 Vt[H][KT];        // value tile, transposed
    __shared__ __align__(16) __bf16 Ps[8][16][KT];    // wave-private P re-layout

    const int tid = threadIdx.x, lane = tid & 31, wid = tid >> 5;
    const int hf = lane >> 4, l16 = lane & 15;
    const int u = blockIdx.y, v = u >> 2;
    const int q0 = blockIdx.x * BR;
    const int qw = q0 + wid * 16;                     // this wave's first q row

    // Q fragments: 3 chunks of K=32 over H=96, loaded straight from global
    bf16x16 aq[3];
    {
        const __bf16* qp = Q + ((size_t)u * S + qw + l16) * H;
#pragma unroll
        for (int c = 0; c < 3; ++c) aq[c] = ld_chunk(qp, c, hf);
    }

    float row_m[8], row_l[8];
#pragma unroll
    for (int r = 0; r < 8; ++r) { row_m[r] = -1e30f; row_l[r] = 0.f; }
    f32x8 oacc[6] = {};

    const float scale = 0.1020620726159658f;          // 1/sqrt(96)
    const int nkt = q0 / KT + 2;                      // causal tile count

    for (int kt = 0; kt < nkt; ++kt) {
        const int kbase = kt * KT;
        // ---- stage K tile: 24 halfs / thread, contiguous
        {
            int row = tid >> 2, c0 = (tid & 3) * 24;
            const __bf16* src = Kh + ((size_t)v * S + kbase + row) * H + c0;
#if USE_ASYNC_LDS
            async_cp_b128(src,      &Ks[row][c0]);
            async_cp_b128(src + 8,  &Ks[row][c0 + 8]);
            async_cp_b128(src + 16, &Ks[row][c0 + 16]);
#else
            const uint4* s4 = (const uint4*)src;
            uint4* dst = (uint4*)&Ks[row][c0];
            dst[0] = s4[0]; dst[1] = s4[1]; dst[2] = s4[2];
#endif
        }
        // ---- stage V tile transposed: Vt[h][key]
        {
            int key = tid & 63, h0 = (tid >> 6) * 24;
            const __bf16* src = Vh + ((size_t)v * S + kbase + key) * H + h0;
#pragma unroll
            for (int j = 0; j < 24; ++j) Vt[h0 + j][key] = src[j];
        }
#if USE_ASYNC_LDS
        wait_async0();
#endif
        __syncthreads();

        if (kbase <= qw + 15) {   // wave-uniform causal skip
            // ---- scores = Q K^T  (B-frag reads Ks rows: K^T in A layout)
            f32x8 sacc[4] = {};
#pragma unroll
            for (int c = 0; c < 3; ++c)
#pragma unroll
                for (int nt = 0; nt < 4; ++nt) {
                    bf16x16 bfr = ld_chunk(&Ks[nt * 16 + l16][0], c, hf);
                    sacc[nt] = __builtin_amdgcn_wmma_f32_16x16x32_bf16(
                        false, aq[c], false, bfr, (short)0, sacc[nt], false, false);
                }
            // ---- mask + scale + row max (16-lane half-wave reduction)
            float p[4][8], mnew[8];
#pragma unroll
            for (int r = 0; r < 8; ++r) {
                int qrow = qw + hf * 8 + r;
                float mx = -1e30f;
#pragma unroll
                for (int nt = 0; nt < 4; ++nt) {
                    int key = kbase + nt * 16 + l16;
                    float vv = sacc[nt][r] * scale;
                    if (key > qrow) vv = -1e9f;
                    p[nt][r] = vv;
                    mx = fmaxf(mx, vv);
                }
#pragma unroll
                for (int m = 1; m < 16; m <<= 1) mx = fmaxf(mx, __shfl_xor(mx, m, 32));
                mnew[r] = fmaxf(row_m[r], mx);
            }
            // ---- exponentiate, rescale running state
#pragma unroll
            for (int r = 0; r < 8; ++r) {
                float alpha = __expf(row_m[r] - mnew[r]);
                float sum = 0.f;
#pragma unroll
                for (int nt = 0; nt < 4; ++nt) {
                    float e = __expf(p[nt][r] - mnew[r]);
                    p[nt][r] = e;
                    sum += e;
                }
#pragma unroll
                for (int m = 1; m < 16; m <<= 1) sum += __shfl_xor(sum, m, 32);
                row_l[r] = row_l[r] * alpha + sum;
                row_m[r] = mnew[r];
#pragma unroll
                for (int ht = 0; ht < 6; ++ht) oacc[ht][r] *= alpha;
            }
            // ---- C-layout -> A-layout via wave-private LDS
#pragma unroll
            for (int nt = 0; nt < 4; ++nt)
#pragma unroll
                for (int r = 0; r < 8; ++r)
                    Ps[wid][hf * 8 + r][nt * 16 + l16] = f2bf16(p[nt][r]);
            asm volatile("s_wait_dscnt 0" ::: "memory");
            // ---- O += P V  (2 key-chunks of 32)
#pragma unroll
            for (int c2 = 0; c2 < 2; ++c2) {
                bf16x16 ap = ld_chunk(&Ps[wid][l16][0], c2, hf);
#pragma unroll
                for (int ht = 0; ht < 6; ++ht) {
                    bf16x16 bv = ld_chunk(&Vt[ht * 16 + l16][0], c2, hf);
                    oacc[ht] = __builtin_amdgcn_wmma_f32_16x16x32_bf16(
                        false, ap, false, bv, (short)0, oacc[ht], false, false);
                }
            }
        }
        __syncthreads();
    }

    // ---- normalize and store bf16 at [row][u*96 + h], row stride 3072
#pragma unroll
    for (int ht = 0; ht < 6; ++ht)
#pragma unroll
        for (int r = 0; r < 8; ++r) {
            int row = qw + hf * 8 + r;
            int col = u * 96 + ht * 16 + l16;
            O[(size_t)row * 3072 + col] = f2bf16(oacc[ht][r] / row_l[r]);
        }
}

// ---------------------------------------------------------------------------
// Host-side orchestration
// ---------------------------------------------------------------------------
extern "C" void kernel_launch(void* const* d_in, const int* in_sizes, int n_in,
                              void* d_out, int out_size, void* d_ws, size_t ws_size,
                              hipStream_t stream) {
    (void)in_sizes; (void)n_in; (void)out_size; (void)ws_size;
    const int S = 2048, M = 3072, U = 32, V = 8, H = 96;
    const int NQ = U * H;    // 3072
    const int NKV = V * H;   // 768

    const float* hidden = (const float*)d_in[0];
    // d_in[1] = attention_mask (causal, handled analytically)
    const float* Wq = (const float*)d_in[2];
    const float* Wk = (const float*)d_in[3];
    const float* Wv = (const float*)d_in[4];
    const float* Wo = (const float*)d_in[5];
    float* out = (float*)d_out;

    char* w = (char*)d_ws;
    auto alloc = [&](size_t bytes) { char* p = w; w += (bytes + 255) & ~(size_t)255; return p; };
    float*  Qf = (float*)alloc((size_t)S * NQ  * 4);
    float*  Kf = (float*)alloc((size_t)S * NKV * 4);
    float*  Vf = (float*)alloc((size_t)S * NKV * 4);
    __bf16* Qr = (__bf16*)alloc((size_t)S * NQ  * 2);   // [u][s][96] post-RoPE
    __bf16* Kr = (__bf16*)alloc((size_t)S * NKV * 2);   // [v][s][96] post-RoPE
    __bf16* Vb = (__bf16*)alloc((size_t)S * NKV * 2);   // [v][s][96]
    __bf16* At = (__bf16*)alloc((size_t)S * NQ  * 2);   // [s][u*96+h]

    dim3 blk(256);
    // QKV projections (fp32 A, bf16 WMMA, fp32 out)
    wmma_gemm<false><<<dim3(NQ  / 128, S / 128), blk, 0, stream>>>(hidden, Wq, Qf, S, NQ,  M);
    wmma_gemm<false><<<dim3(NKV / 128, S / 128), blk, 0, stream>>>(hidden, Wk, Kf, S, NKV, M);
    wmma_gemm<false><<<dim3(NKV / 128, S / 128), blk, 0, stream>>>(hidden, Wv, Vf, S, NKV, M);

    // RoPE + layout change + bf16
    int tq = S * U * 48;
    rope_to_bf16<<<(tq + 255) / 256, blk, 0, stream>>>(Qf, Qr, U, S);
    int tk = S * V * 48;
    rope_to_bf16<<<(tk + 255) / 256, blk, 0, stream>>>(Kf, Kr, V, S);
    int tv = S * V * 96;
    conv_to_bf16_hm<<<(tv + 255) / 256, blk, 0, stream>>>(Vf, Vb, V, S);

    // Flash attention, one block per (q-tile, head)
    flash_attn<<<dim3(S / 128, U), blk, 0, stream>>>(Qr, Kr, Vb, At, S);

    // Output projection: attn(bf16) x Wo(fp32) -> fp32 out
    wmma_gemm<true><<<dim3(M / 128, S / 128), blk, 0, stream>>>(At, Wo, out, S, M, NQ);
}
